// SelfAttention_61641370632288
// MI455X (gfx1250) — matrix operand order repaired and checked by reading, not compiled, not run on hardware
//
#include <hip/hip_runtime.h>
#include <hip/hip_bf16.h>

// ---------------------------------------------------------------------------
// B=8, C=256, Cqk=16, N=4096.
// Workspace (f16 unless noted):
//   xt  : [B][N][C]  @ 0        (16 MB)  pixel-major x, f16
//   qh  : [B][N][16] @ 16 MB    (1 MB)   q * log2(e), row per pixel
//   kh  : [B][N][16] @ 17 MB    (1 MB)
//   vt  : [B][C][N]  @ 18 MB    (16 MB)  channel-major v (key dim contiguous)
//   wqh : [16][256]  @ 34 MB    (8 KB)
//   wkh : [16][256]            (8 KB)
//   wvh : [256][256]           (128 KB)
// ---------------------------------------------------------------------------

typedef __attribute__((ext_vector_type(16))) _Float16 v16h;
typedef __attribute__((ext_vector_type(8)))  _Float16 v8h;
typedef __attribute__((ext_vector_type(8)))  float    v8f;

static __device__ __forceinline__ v16h cat8(v8h lo, v8h hi) {
  return __builtin_shufflevector(lo, hi, 0, 1, 2, 3, 4, 5, 6, 7,
                                         8, 9, 10, 11, 12, 13, 14, 15);
}

// A-operand tile loader (16xK32 f16): rows of a row-major [*][256] matrix.
// lanes 0-15: e0..7 = K kk..kk+7, e8..15 = kk+16..23; lanes>=16: +8.
static __device__ __forceinline__ v16h loadA(const _Float16* base, long row,
                                             int kk, bool hi) {
  const _Float16* p = base + row * 256 + kk + (hi ? 8 : 0);
  return cat8(*(const v8h*)p, *(const v8h*)(p + 16));
}
// B-operand tile loader (K32x16 f16): "column" = one row of a row-major
// [*][256] matrix. lanes 0-15: e0..7 = K kk..kk+7, e8..15 = kk+16..23; lanes>=16: +8.
static __device__ __forceinline__ v16h loadB(const _Float16* base, long row,
                                             int kk, bool hi) {
  const _Float16* p = base + row * 256 + kk + (hi ? 8 : 0);
  return cat8(*(const v8h*)p, *(const v8h*)(p + 16));
}

// DPP16 ROW_XMASK butterfly (pure VALU, no DScnt round-trips).
template <int MASK>
static __device__ __forceinline__ float xmax16(float v) {
  int s = __builtin_bit_cast(int, v);
  int p = __builtin_amdgcn_update_dpp(s, s, 0x160 | MASK, 0xf, 0xf, false);
  return fmaxf(v, __builtin_bit_cast(float, p));
}
static __device__ __forceinline__ float rowmax16(float v) {
  v = xmax16<1>(v); v = xmax16<2>(v); v = xmax16<4>(v); v = xmax16<8>(v);
  return v;
}

#define BATCH 8
#define CH    256
#define CQK   16
#define NPIX  4096
#define LOG2E 1.44269504088896340736f

// ---------------------------------------------------------------------------
// Kernel 0a: convert Wq/Wk/Wv to f16 workspace copies.
// ---------------------------------------------------------------------------
__global__ __launch_bounds__(256) void cvt_weights_kernel(
    const float* __restrict__ Wq, const float* __restrict__ Wk,
    const float* __restrict__ Wv, _Float16* __restrict__ wqh,
    _Float16* __restrict__ wkh, _Float16* __restrict__ wvh) {
  const int idx = blockIdx.x * 256 + threadIdx.x;
  if (idx < CQK * CH) {
    wqh[idx] = (_Float16)(Wq[idx] * LOG2E);  // fold log2(e) into q path
    wkh[idx] = (_Float16)Wk[idx];
  }
  if (idx < CH * CH) wvh[idx] = (_Float16)Wv[idx];
}

// ---------------------------------------------------------------------------
// Kernel 0b: x [B][C][N] f32 -> xt [B][N][C] f16 (LDS-tiled transpose).
// grid (N/32, C/32, B), block (32,8).
// ---------------------------------------------------------------------------
__global__ __launch_bounds__(256) void transpose_x_kernel(
    const float* __restrict__ x, _Float16* __restrict__ xt) {
  __shared__ float tile[32][33];
  const int b = blockIdx.z, c0 = blockIdx.y * 32, n0 = blockIdx.x * 32;
  const int tx = threadIdx.x, ty = threadIdx.y;
#pragma unroll
  for (int i = 0; i < 4; ++i) {
    const int c = ty + 8 * i;
    tile[c][tx] = x[((long)b * CH + c0 + c) * NPIX + n0 + tx];
  }
  __syncthreads();
#pragma unroll
  for (int i = 0; i < 4; ++i) {
    const int n = ty + 8 * i;
    xt[((long)b * NPIX + n0 + n) * CH + c0 + tx] = (_Float16)tile[tx][n];
  }
}

// ---------------------------------------------------------------------------
// Kernel 1: WMMA QKV projection. grid (N/64, B), block 128 (4 waves).
// Each wave owns 16 pixels: Dq/Dk = W x X (D rows = channels),
// Dv = X^T x Wv^T (D rows = pixels -> contiguous stores into [C][N]).
// ---------------------------------------------------------------------------
__global__ __launch_bounds__(128) void wmma_proj_kernel(
    const _Float16* __restrict__ xt, const _Float16* __restrict__ wqh,
    const _Float16* __restrict__ wkh, const _Float16* __restrict__ wvh,
    const float* __restrict__ bq, const float* __restrict__ bk,
    const float* __restrict__ bv, _Float16* __restrict__ qh,
    _Float16* __restrict__ kh, _Float16* __restrict__ vt) {
  const int b    = blockIdx.y;
  const int wave = threadIdx.x >> 5;
  const int lane = threadIdx.x & 31;
  const int lh   = lane & 15;
  const bool hi  = lane >= 16;
  const int n0   = blockIdx.x * 64 + wave * 16;
  const long xrow0 = (long)b * NPIX + n0;

  v8f Dq = {}, Dk = {};
  v8f Dv[16];
#pragma unroll
  for (int ct = 0; ct < 16; ++ct) Dv[ct] = (v8f){};

  for (int kk = 0; kk < CH; kk += 32) {
    const v16h aX  = loadA(xt, xrow0 + lh, kk, hi);   // X^T tile (M=pixels)
    const v16h bX  = loadB(xt, xrow0 + lh, kk, hi);   // X tile   (N=pixels)
    const v16h aWq = loadA(wqh, lh, kk, hi);
    const v16h aWk = loadA(wkh, lh, kk, hi);
    Dq = __builtin_amdgcn_wmma_f32_16x16x32_f16(false, aWq, false, bX,
                                                (short)0, Dq, false, false);
    Dk = __builtin_amdgcn_wmma_f32_16x16x32_f16(false, aWk, false, bX,
                                                (short)0, Dk, false, false);
#pragma unroll
    for (int ct = 0; ct < 16; ++ct) {
      const v16h bWv = loadB(wvh, ct * 16 + lh, kk, hi);  // Wv^T (N=out ch)
      Dv[ct] = __builtin_amdgcn_wmma_f32_16x16x32_f16(
          false, aX, false, bWv, (short)0, Dv[ct], false, false);
    }
  }

  // q/k: lane = pixel n0+lh, r = out channel (hi ? 8+r : r). Contiguous b128.
  {
    const int ob = hi ? 8 : 0;
    v8h pq, pk;
#pragma unroll
    for (int r = 0; r < 8; ++r) {
      pq[r] = (_Float16)((Dq[r] + bq[ob + r] ) * LOG2E);
      pk[r] = (_Float16)(Dk[r] + bk[ob + r]);
    }
    _Float16* qp = qh + ((long)b * NPIX + n0 + lh) * CQK + ob;
    _Float16* kp = kh + ((long)b * NPIX + n0 + lh) * CQK + ob;
    *(v8h*)qp = pq;
    *(v8h*)kp = pk;
  }
  // v: lane = out channel ct*16+lh, r = pixel n0+(hi?8:0)+r. Contiguous b128.
#pragma unroll
  for (int ct = 0; ct < 16; ++ct) {
    const int c = ct * 16 + lh;
    const float bias = bv[c];
    v8h pv;
#pragma unroll
    for (int r = 0; r < 8; ++r) pv[r] = (_Float16)(Dv[ct][r] + bias);
    *(v8h*)(vt + ((long)b * CH + c) * NPIX + n0 + (hi ? 8 : 0)) = pv;
  }
}

// ---------------------------------------------------------------------------
// Kernel 2: flash attention + residual. grid (N/64, B), block 128 (4 waves).
// Row-max via DPP16 ROW_XMASK; row-sum via WMMA against all-ones B matrix.
// ---------------------------------------------------------------------------
__global__ __launch_bounds__(128) void flash_attn_kernel(
    const _Float16* __restrict__ qh, const _Float16* __restrict__ kh,
    const _Float16* __restrict__ vt, const float* __restrict__ x,
    const float* __restrict__ gptr, float* __restrict__ out) {
  __shared__ __align__(16) _Float16 Plds[4][16][40];  // padded banks

  const int b    = blockIdx.y;
  const int wave = threadIdx.x >> 5;
  const int lane = threadIdx.x & 31;
  const int lh   = lane & 15;
  const bool hi  = lane >= 16;
  const int i0   = blockIdx.x * 64 + wave * 16;

  const v8h z8 = {};
  v16h ones;
#pragma unroll
  for (int i = 0; i < 16; ++i) ones[i] = (_Float16)1.0f;

  // A operand: q tile 16x32, K=16..31 zero padded (q pre-scaled by log2e).
  const v8h* qp = (const v8h*)(qh + ((long)b * NPIX + i0 + lh) * CQK);
  const v16h aQ = hi ? cat8(z8, z8) : cat8(qp[0], qp[1]);

  float mrow[8], lrow[8];
#pragma unroll
  for (int r = 0; r < 8; ++r) { mrow[r] = -__builtin_inff(); lrow[r] = 0.0f; }
  v8f acc[16];
#pragma unroll
  for (int ct = 0; ct < 16; ++ct) acc[ct] = (v8f){};

  for (int jb = 0; jb < NPIX / 32; ++jb) {
    const int j0 = jb * 32;

    v16h bK0 = cat8(z8, z8), bK1 = cat8(z8, z8);
    if (!hi) {
      const v8h* kp0 = (const v8h*)(kh + ((long)b * NPIX + j0 + lh) * CQK);
      const v8h* kp1 = (const v8h*)(kh + ((long)b * NPIX + j0 + 16 + lh) * CQK);
      bK0 = cat8(kp0[0], kp0[1]);
      bK1 = cat8(kp1[0], kp1[1]);
      if (jb + 1 < NPIX / 32)
        __builtin_prefetch(kh + ((long)b * NPIX + j0 + 32 + lh) * CQK, 0, 0);
    }

    const v8f zc = {};
    v8f S0 = __builtin_amdgcn_wmma_f32_16x16x32_f16(false, aQ, false, bK0,
                                                    (short)0, zc, false, false);
    v8f S1 = __builtin_amdgcn_wmma_f32_16x16x32_f16(false, aQ, false, bK1,
                                                    (short)0, zc, false, false);

    // pass 1: row max (DPP), rescale, exp2, stash P (f16) in LDS.
#pragma unroll
    for (int r = 0; r < 8; ++r) {
      const float s0 = S0[r], s1 = S1[r];
      const float t  = rowmax16(fmaxf(s0, s1));
      const float mn = fmaxf(mrow[r], t);
      const float sc = exp2f(mrow[r] - mn);
      mrow[r] = mn;
      lrow[r] *= sc;
      const float p0 = exp2f(s0 - mn);
      const float p1 = exp2f(s1 - mn);
#pragma unroll
      for (int ct = 0; ct < 16; ++ct) acc[ct][r] *= sc;
      const int prow = r + (hi ? 8 : 0);
      Plds[wave][prow][lh]      = (_Float16)p0;
      Plds[wave][prow][lh + 16] = (_Float16)p1;
    }

    asm volatile("s_wait_dscnt 0" ::: "memory");
    const int ka = hi ? 8 : 0;
    const v8h plo = *(const v8h*)&Plds[wave][lh][ka];
    const v8h phi = *(const v8h*)&Plds[wave][lh][16 + ka];
    const v16h aP = cat8(plo, phi);
    asm volatile("" ::: "memory");

    // row sums of the f16 P actually used for O: one WMMA against ones.
    const v8f RS = __builtin_amdgcn_wmma_f32_16x16x32_f16(
        false, aP, false, ones, (short)0, zc, false, false);
#pragma unroll
    for (int r = 0; r < 8; ++r) lrow[r] += RS[r];

    // O += P * V : 16 channel tiles, K=32 keys.
#pragma unroll
    for (int ct = 0; ct < 16; ++ct) {
      const int c = ct * 16 + lh;
      const v8h* vp = (const v8h*)(vt + ((long)b * CH + c) * NPIX + j0 +
                                   (hi ? 8 : 0));
      const v16h bV = cat8(vp[0], vp[2]);
      acc[ct] = __builtin_amdgcn_wmma_f32_16x16x32_f16(
          false, aP, false, bV, (short)0, acc[ct], false, false);
    }
  }

  const float gamma = *gptr;
  float invl[8];
#pragma unroll
  for (int r = 0; r < 8; ++r) invl[r] = 1.0f / lrow[r];

#pragma unroll
  for (int ct = 0; ct < 16; ++ct) {
    const int c = ct * 16 + lh;
    const long base = ((long)b * CH + c) * NPIX + i0 + (hi ? 8 : 0);
#pragma unroll
    for (int r = 0; r < 8; ++r) {
      const long idx = base + r;
      out[idx] = x[idx] + gamma * acc[ct][r] * invl[r];
    }
  }
}

// ---------------------------------------------------------------------------
extern "C" void kernel_launch(void* const* d_in, const int* in_sizes, int n_in,
                              void* d_out, int out_size, void* d_ws,
                              size_t ws_size, hipStream_t stream) {
  (void)in_sizes; (void)n_in; (void)out_size; (void)ws_size;
  const float* x  = (const float*)d_in[0];
  const float* Wq = (const float*)d_in[1];
  const float* bq = (const float*)d_in[2];
  const float* Wk = (const float*)d_in[3];
  const float* bk = (const float*)d_in[4];
  const float* Wv = (const float*)d_in[5];
  const float* bv = (const float*)d_in[6];
  const float* gm = (const float*)d_in[7];
  float* out = (float*)d_out;

  char* ws = (char*)d_ws;
  const size_t MB = 1u << 20;
  _Float16* xt  = (_Float16*)(ws);              // 16 MB
  _Float16* qh  = (_Float16*)(ws + 16 * MB);    // 1 MB
  _Float16* kh  = (_Float16*)(ws + 17 * MB);    // 1 MB
  _Float16* vt  = (_Float16*)(ws + 18 * MB);    // 16 MB
  _Float16* wqh = (_Float16*)(ws + 34 * MB);    // 8 KB
  _Float16* wkh = (_Float16*)(ws + 34 * MB + 8192);
  _Float16* wvh = (_Float16*)(ws + 34 * MB + 16384);  // 128 KB

  cvt_weights_kernel<<<CH * CH / 256, 256, 0, stream>>>(Wq, Wk, Wv, wqh, wkh,
                                                        wvh);
  transpose_x_kernel<<<dim3(NPIX / 32, CH / 32, BATCH), dim3(32, 8), 0,
                       stream>>>(x, xt);
  wmma_proj_kernel<<<dim3(NPIX / 64, BATCH), 128, 0, stream>>>(
      xt, wqh, wkh, wvh, bq, bk, bv, qh, kh, vt);
  flash_attn_kernel<<<dim3(NPIX / 64, BATCH), 128, 0, stream>>>(qh, kh, vt, x,
                                                                gm, out);
}
